// CalibLoss_47175920779952
// MI455X (gfx1250) — compile-verified, matrix operand order checked by reading, not run on hardware
//
#include <hip/hip_runtime.h>

#define S_SP   (96*192*192)          // 3,538,944 spatial elems per batch item
#define NPOS   (2*S_SP)              // 7,077,888 positions
#define NGROUP (NPOS/4)              // float4 groups = 1,769,472
#define NBINS  15
#define NCLS   6                     // classes 1..6
#define REP    8                     // LDS histogram replication (conflict <= 4)
#define HWORDS (2*NCLS*NBINS*REP)    // 1440 u32
#define NCOUNT (2*NCLS*NBINS)        // 180
#define CE_SCALE 67108864.0f         // 2^26 fixed point for deterministic CE sum

typedef __attribute__((ext_vector_type(2)))  float    v2f;
typedef __attribute__((ext_vector_type(8)))  float    v8f;
typedef __attribute__((ext_vector_type(16))) _Float16 v16h;

__device__ __forceinline__ float f4e(const float4& v, int p) {
  return p == 0 ? v.x : p == 1 ? v.y : p == 2 ? v.z : v.w;
}
__device__ __forceinline__ int i4e(const int4& v, int p) {
  return p == 0 ? v.x : p == 1 ? v.y : p == 2 ? v.z : v.w;
}
__device__ __forceinline__ float sigf(float v) {
  return 1.0f / (1.0f + __expf(-v));
}
__device__ __forceinline__ void wait_async0() {
#if __has_builtin(__builtin_amdgcn_s_wait_asynccnt)
  __builtin_amdgcn_s_wait_asynccnt(0);
#else
  asm volatile("s_wait_asynccnt 0x0" ::: "memory");
#endif
}

// ---------------------------------------------------------------- init
__global__ void calib_init(unsigned* ws) {
  const int i = threadIdx.x;
  if (i < NCOUNT + 2) ws[i] = 0u;   // [0..1]=u64 ce, [2..181]=counters
}

// ---------------------------------------------------------------- main pass
__global__ __launch_bounds__(256) void calib_main(
    const float* __restrict__ x, const int* __restrict__ y,
    unsigned long long* __restrict__ g_ce,
    unsigned* __restrict__ g_total, unsigned* __restrict__ g_true)
{
  __shared__ unsigned hist[HWORDS];
  __shared__ int4 ybuf[256];
  __shared__ unsigned long long ce_sh;
  const int tid = threadIdx.x;
  for (int i = tid; i < HWORDS; i += 256) hist[i] = 0u;
  if (tid == 0) ce_sh = 0ull;
  __syncthreads();

  const int rep    = tid & (REP - 1);
  const int stride = (int)gridDim.x * 256;
  float ce = 0.0f;

  for (int g = (int)blockIdx.x * 256 + tid; g < NGROUP; g += stride) {
    const int gs = g << 2;
    const int b  = (gs >= S_SP) ? 1 : 0;
    const int s  = gs - b * S_SP;
    const float* xp = x + b * 8 * S_SP + s;
    const int*   yp = y + b * S_SP + s;

    // CDNA5 async DMA: y tile -> LDS (tracked by ASYNCcnt)
    {
      unsigned            ldso = (unsigned)(size_t)&ybuf[tid];
      unsigned long long  ga   = (unsigned long long)(size_t)yp;
      asm volatile("global_load_async_to_lds_b128 %0, %1, off"
                   :: "v"(ldso), "v"(ga) : "memory");
    }

    float4 xv[8];
#pragma unroll
    for (int c = 0; c < 8; ++c)
      xv[c] = *(const float4*)(xp + c * S_SP);

    // Prefetch next grid-stride tile into cache (global_prefetch_b8)
    const int gn = g + stride;
    if (gn < NGROUP) {
      const int gsn = gn << 2;
      const int bn  = (gsn >= S_SP) ? 1 : 0;
      const int sn  = gsn - bn * S_SP;
      const float* xpn = x + bn * 8 * S_SP + sn;
#pragma unroll
      for (int c = 0; c < 8; ++c) __builtin_prefetch(xpn + c * S_SP, 0, 0);
      __builtin_prefetch(y + bn * S_SP + sn, 0, 0);
    }

    wait_async0();
    const int4 yv = ybuf[tid];

#pragma unroll
    for (int p = 0; p < 4; ++p) {
      float v[8];
#pragma unroll
      for (int c = 0; c < 8; ++c) v[c] = f4e(xv[c], p);
      float m = v[0];
#pragma unroll
      for (int c = 1; c < 8; ++c) m = fmaxf(m, v[c]);
      float e[8];
      float sum = 0.0f;
#pragma unroll
      for (int c = 0; c < 8; ++c) { e[c] = __expf(v[c] - m); sum += e[c]; }
      const int tgt = i4e(yv, p);
      float vt = v[0];
#pragma unroll
      for (int c = 1; c < 8; ++c) vt = (tgt == c) ? v[c] : vt;
      ce += m + __logf(sum) - vt;              // -log softmax[target]
      const float inv = __builtin_amdgcn_rcpf(sum);
#pragma unroll
      for (int j = 0; j < NCLS; ++j) {         // classes 1..6
        const float prob = e[j + 1] * inv;
        int bin = (int)(prob * 15.0f);         // 1+1e-8 == 1 in f32
        bin = bin > 14 ? 14 : bin;
        atomicAdd(&hist[(j * NBINS + bin) * REP + rep], 1u);
        if (tgt == j + 1)
          atomicAdd(&hist[((NCLS + j) * NBINS + bin) * REP + rep], 1u);
      }
    }
  }

  // deterministic fixed-point CE reduction: thread -> LDS u64 -> global u64
  const unsigned long long cefx =
      (unsigned long long)(long long)(ce * CE_SCALE + 0.5f);
  atomicAdd(&ce_sh, cefx);
  __syncthreads();

  for (int i = tid; i < NCOUNT; i += 256) {
    unsigned t = 0;
#pragma unroll
    for (int r = 0; r < REP; ++r) t += hist[i * REP + r];
    if (t) {
      if (i < NCLS * NBINS) atomicAdd(&g_total[i], t);
      else                  atomicAdd(&g_true[i - NCLS * NBINS], t);
    }
  }
  if (tid == 0) atomicAdd(g_ce, ce_sh);
}

// ---------------------------------------------------------------- finalize
// One wave. Packs the 90 per-(class,bin) squared diffs into WMMA A-tiles and
// reduces with B = ones: D[m][n] = sum_k A[m][k] (n-invariant), so summing the
// 8 D VGPRs + one cross-half shuffle gives the exact total independent of the
// A slot mapping.
__global__ __launch_bounds__(32) void calib_fin(
    const float* __restrict__ calib,
    const unsigned long long* __restrict__ g_ce,
    const unsigned* __restrict__ g_total,
    const unsigned* __restrict__ g_true,
    float* __restrict__ out)
{
  const int lane = threadIdx.x;
  float vals[3];
#pragma unroll
  for (int k = 0; k < 3; ++k) {
    const int i = lane + 32 * k;
    float v = 0.0f;
    if (i < NCLS * NBINS) {
      const int j = i / NBINS;          // class index 0..5 (class = j+1)
      const int b = i - j * NBINS;      // bin 0..14
      const float ratio = sigf((float)g_true[i]) / sigf((float)g_total[i]);
      const float sc    = sigf(calib[b * 8 + (j + 1)]);
      const float d     = sc - ratio;
      v = d * d * (1.0f / 15.0f);       // mean over bins folded in
    }
    vals[k] = v;
  }

  float s;
#if __has_builtin(__builtin_amdgcn_wmma_f32_16x16x4_f32)
  {
    v2f a0 = { vals[0], vals[1] };
    v2f a1 = { vals[2], 0.0f };
    v2f bo = { 1.0f, 1.0f };
    v8f c  = {};
    c = __builtin_amdgcn_wmma_f32_16x16x4_f32(false, a0, false, bo,
                                              (short)0, c, false, false);
    c = __builtin_amdgcn_wmma_f32_16x16x4_f32(false, a1, false, bo,
                                              (short)0, c, false, false);
    s = c[0] + c[1] + c[2] + c[3] + c[4] + c[5] + c[6] + c[7];
  }
#elif __has_builtin(__builtin_amdgcn_wmma_f32_16x16x32_f16)
  {
    v16h a = {};
    a[0] = (_Float16)vals[0];
    a[1] = (_Float16)vals[1];
    a[2] = (_Float16)vals[2];
    v16h bo;
#pragma unroll
    for (int i = 0; i < 16; ++i) bo[i] = (_Float16)1.0f;
    v8f c = {};
    c = __builtin_amdgcn_wmma_f32_16x16x32_f16(false, a, false, bo,
                                               (short)0, c, false, false);
    s = c[0] + c[1] + c[2] + c[3] + c[4] + c[5] + c[6] + c[7];
  }
#else
  {
    s = vals[0] + vals[1] + vals[2];
#pragma unroll
    for (int off = 16; off > 0; off >>= 1) s += __shfl_down(s, off, 32);
    s = __shfl(s, 0, 32);
  }
#endif
  s += __shfl_down(s, 16, 32);   // rows 0-7 half + rows 8-15 half -> lane 0
  if (lane == 0) {
    const double ce_mean =
        (double)*g_ce * (1.0 / 67108864.0) / (double)NPOS;
    out[0] = (float)ce_mean + s;
  }
}

// ---------------------------------------------------------------- launch
extern "C" void kernel_launch(void* const* d_in, const int* in_sizes, int n_in,
                              void* d_out, int out_size, void* d_ws, size_t ws_size,
                              hipStream_t stream) {
  (void)in_sizes; (void)n_in; (void)out_size; (void)ws_size;
  const float* x     = (const float*)d_in[0];
  const int*   y     = (const int*)d_in[1];
  const float* calib = (const float*)d_in[2];
  float*       out   = (float*)d_out;

  unsigned long long* g_ce = (unsigned long long*)d_ws;   // offset 0, 8B aligned
  unsigned* g_total = (unsigned*)d_ws + 2;                // 90 counters
  unsigned* g_true  = g_total + NCLS * NBINS;             // 90 counters

  calib_init<<<1, 256, 0, stream>>>((unsigned*)d_ws);
  calib_main<<<2048, 256, 0, stream>>>(x, y, g_ce, g_total, g_true);
  calib_fin<<<1, 32, 0, stream>>>(calib, g_ce, g_total, g_true, out);
}
